// PointConvBase_2869038153774
// MI455X (gfx1250) — matrix operand order repaired and checked by reading, not compiled, not run on hardware
//
#include <hip/hip_runtime.h>
#include <hip/hip_bf16.h>

// ---------------- problem constants ----------------
#define BB    4
#define NN    4096
#define SS    4096
#define KK    32
#define CHIN  64
#define CHOUT 64
#define CICM  16
#define PTOT  (BB * SS * KK)      // 524288 points through WeightNet
#define EPSBN 1e-5f

typedef __attribute__((ext_vector_type(16))) _Float16 v16h;
typedef __attribute__((ext_vector_type(8)))  _Float16 v8h;
typedef __attribute__((ext_vector_type(8)))  float    v8f;

// ---------------- ws layout (bytes) ----------------
// idx      : BB*SS*KK int32            = 2,097,152
// stats    : 256 floats (sum/sumsq per layer, stride 64 per layer)
// ss       : 256 floats (scale/shift per layer)
// H3       : PTOT*16 f16               = 16,777,216
// partial  : BB*SS*1024 f16            = 33,554,432
// bpack    : 32*4*32*16 f16            = 131,072
#define OFF_IDX     0
#define OFF_STATS   2097152
#define OFF_SS      (OFF_STATS + 1024)
#define OFF_H3      (OFF_STATS + 2048)
#define OFF_PARTIAL (OFF_H3 + 16777216)
#define OFF_BPACK   (OFF_PARTIAL + 33554432)

// ---------------------------------------------------------------------------
// zero the stats + scale/shift region (graph replays re-accumulate atomics)
// ---------------------------------------------------------------------------
__global__ void k_init_stats(float* stats) {
    if (threadIdx.x < 512) stats[threadIdx.x] = 0.0f;
}

// ---------------------------------------------------------------------------
// KNN: one wave (32 lanes) per query. Distances cached in LDS (16KB),
// 32 rounds of wave-reduced argmin.
// ---------------------------------------------------------------------------
__global__ __launch_bounds__(32) void k_knn(const float* __restrict__ inp_xyz,
                                            const float* __restrict__ query_xyz,
                                            int* __restrict__ idx_out) {
    __shared__ float dist[NN];
    const int q    = blockIdx.x;          // 0 .. BB*SS-1
    const int b    = q / SS;
    const int lane = threadIdx.x;

    const float* qp = query_xyz + (size_t)q * 3;
    const float qx = qp[0], qy = qp[1], qz = qp[2];
    const float* pts = inp_xyz + (size_t)b * NN * 3;

    for (int n = lane; n < NN; n += 32) {
        float dx = pts[n * 3 + 0] - qx;
        float dy = pts[n * 3 + 1] - qy;
        float dz = pts[n * 3 + 2] - qz;
        dist[n] = dx * dx + dy * dy + dz * dz;
    }
    __syncthreads();

    for (int r = 0; r < KK; ++r) {
        float best = 3.402823466e38f;
        int   bi   = 0x7fffffff;
        for (int n = lane; n < NN; n += 32) {
            float d = dist[n];
            if (d < best) { best = d; bi = n; }
        }
        // wave32 min-reduce with index tie-break
        #pragma unroll
        for (int off = 16; off > 0; off >>= 1) {
            float ob = __shfl_xor(best, off, 32);
            int   oi = __shfl_xor(bi,   off, 32);
            if (ob < best || (ob == best && oi < bi)) { best = ob; bi = oi; }
        }
        if (lane == 0) {
            idx_out[(size_t)q * KK + r] = bi;
            dist[bi] = 3.402823466e38f;   // knock out winner
        }
        __syncthreads();
    }
}

// ---------------------------------------------------------------------------
// WeightNet pass. stage = 1/2/3: compute layer-<stage> pre-activation and
// accumulate per-channel sum/sumsq (LDS reduce -> global atomics).
// stage = 4: compute final h3 and store as f16.
// ---------------------------------------------------------------------------
__global__ __launch_bounds__(256) void k_wn(int stage,
                                            const int*   __restrict__ idx,
                                            const float* __restrict__ inp_xyz,
                                            const float* __restrict__ query_xyz,
                                            const float* __restrict__ w1, const float* __restrict__ b1,
                                            const float* __restrict__ w2, const float* __restrict__ b2,
                                            const float* __restrict__ w3, const float* __restrict__ b3,
                                            const float* __restrict__ ss,     // scale/shift, 256 floats
                                            float*       __restrict__ stats,  // sums, 256 floats
                                            _Float16*    __restrict__ H3) {
    __shared__ float sW1[96], sB1[32], sW2[1024], sB2[32], sW3[512], sB3[16];
    __shared__ float sSS[160];
    __shared__ float sAcc[64];

    const int tid = threadIdx.x;
    for (int i = tid; i < 96;   i += 256) sW1[i] = w1[i];
    for (int i = tid; i < 1024; i += 256) sW2[i] = w2[i];
    for (int i = tid; i < 512;  i += 256) sW3[i] = w3[i];
    for (int i = tid; i < 160;  i += 256) sSS[i] = ss[i];
    if (tid < 32) { sB1[tid] = b1[tid]; sB2[tid] = b2[tid]; }
    if (tid < 16) { sB3[tid] = b3[tid]; }
    if (tid < 64) { sAcc[tid] = 0.0f; }
    __syncthreads();

    const int p = blockIdx.x * 256 + tid;   // PTOT divisible by 256
    const int q = p / KK;
    const int b = q / SS;
    const int n = idx[p];

    const float* pt = inp_xyz + ((size_t)b * NN + n) * 3;
    const float* qp = query_xyz + (size_t)q * 3;
    const float d0 = pt[0] - qp[0];
    const float d1 = pt[1] - qp[1];
    const float d2 = pt[2] - qp[2];

    float y[32], h[32];

    // layer 1: 3 -> 32
    #pragma unroll
    for (int o = 0; o < 32; ++o)
        y[o] = sW1[o * 3 + 0] * d0 + sW1[o * 3 + 1] * d1 + sW1[o * 3 + 2] * d2 + sB1[o];

    int nch = 32;
    if (stage >= 2) {
        #pragma unroll
        for (int o = 0; o < 32; ++o)
            h[o] = fmaxf(0.0f, y[o] * sSS[o] + sSS[32 + o]);
        // layer 2: 32 -> 32
        #pragma unroll
        for (int o = 0; o < 32; ++o) {
            float acc = sB2[o];
            #pragma unroll
            for (int i = 0; i < 32; ++i) acc += sW2[o * 32 + i] * h[i];
            y[o] = acc;
        }
    }
    if (stage >= 3) {
        #pragma unroll
        for (int o = 0; o < 32; ++o)
            h[o] = fmaxf(0.0f, y[o] * sSS[64 + o] + sSS[96 + o]);
        // layer 3: 32 -> 16
        #pragma unroll
        for (int o = 0; o < 16; ++o) {
            float acc = sB3[o];
            #pragma unroll
            for (int i = 0; i < 32; ++i) acc += sW3[o * 32 + i] * h[i];
            y[o] = acc;
        }
        nch = 16;
    }

    if (stage <= 3) {
        // accumulate per-channel sum / sumsq for BatchNorm
        for (int o = 0; o < nch; ++o) {
            atomicAdd(&sAcc[o],      y[o]);
            atomicAdd(&sAcc[32 + o], y[o] * y[o]);
        }
        __syncthreads();
        float* gbase = stats + (stage - 1) * 64;
        if (tid < nch) {
            atomicAdd(&gbase[tid],      sAcc[tid]);
            atomicAdd(&gbase[32 + tid], sAcc[32 + tid]);
        }
    } else {
        // stage 4: apply BN3 + relu, emit h3 as f16 [PTOT][16]
        #pragma unroll
        for (int o = 0; o < 16; ++o) {
            float v = fmaxf(0.0f, y[o] * sSS[128 + o] + sSS[144 + o]);
            H3[(size_t)p * 16 + o] = (_Float16)v;
        }
    }
}

// ---------------------------------------------------------------------------
// finalize BN stats for one layer: scale = g*rsqrt(var+eps), shift = be - m*scale
// ---------------------------------------------------------------------------
__global__ void k_finalize(const float* __restrict__ sums, float* __restrict__ ssout,
                           const float* __restrict__ g, const float* __restrict__ be, int nch) {
    int o = threadIdx.x;
    if (o >= nch) return;
    const float inv = 1.0f / (float)PTOT;
    float m  = sums[o] * inv;
    float var = sums[32 + o] * inv - m * m;
    float sc = g[o] * rsqrtf(var + EPSBN);
    ssout[o]       = sc;
    ssout[nch + o] = be[o] - m * sc;
}

// ---------------------------------------------------------------------------
// pack lw (CHOUT x 1024, row-major) into the per-lane WMMA B-operand layout:
// bpack[kk][nt][lane][j], lane<16: B[K=kk*32+j+(j&8)][N=nt*16+lane], lane>=16: K+=8
// so the GEMM reads two contiguous b128 per lane.
// ---------------------------------------------------------------------------
__global__ void k_pack_lw(const float* __restrict__ lw, _Float16* __restrict__ bp) {
    int t = blockIdx.x * blockDim.x + threadIdx.x;   // < 65536
    int j    = t & 15;
    int lane = (t >> 4) & 31;
    int nt   = (t >> 9) & 3;
    int kk   = t >> 11;
    int Kidx = kk * 32 + ((lane & 16) ? 8 : 0) + j + (j & 8);
    int Nidx = nt * 16 + (lane & 15);
    bp[t] = (_Float16)lw[Nidx * 1024 + Kidx];
}

// ---------------------------------------------------------------------------
// Aggregation: per query, partial[c,o] = sum_k vals[k,c]*h3[k,o]
// = 4 tiles of V_WMMA_F32_16X16X32_F16 (A = vals 16x32, B = h3 32x16).
// One wave per query; EXEC all-ones at WMMA.
// ---------------------------------------------------------------------------
__global__ __launch_bounds__(32) void k_aggregate(const int* __restrict__ idx,
                                                  const float* __restrict__ inp_vals,
                                                  const _Float16* __restrict__ H3,
                                                  _Float16* __restrict__ partial) {
    __shared__ float sVals[KK][CHIN + 1];    // +1 pad vs bank conflicts
    const int q    = blockIdx.x;
    const int b    = q / SS;
    const int lane = threadIdx.x;

    const int n = idx[(size_t)q * KK + lane];
    const float* vp = inp_vals + ((size_t)b * NN + n) * CHIN;
    #pragma unroll 4
    for (int c = 0; c < CHIN; ++c) sVals[lane][c] = vp[c];
    __syncthreads();

    // B operand: B[K=k][N=o] = h3[k][o]
    const _Float16* h3 = H3 + (size_t)q * KK * 16;
    const int col = lane & 15;
    const int kb  = (lane & 16) ? 8 : 0;
    v16h bm;
    #pragma unroll
    for (int j = 0; j < 16; ++j) bm[j] = h3[(kb + j + (j & 8)) * 16 + col];

    const int koff  = (lane & 16) ? 8 : 0;   // A-lane K sets {0..7,16..23} / {8..15,24..31}
    const int mbase = (lane & 16) ? 8 : 0;   // C row offset

    for (int t = 0; t < 4; ++t) {
        const int c = t * 16 + col;          // A row M = channel within tile
        v16h am;
        #pragma unroll
        for (int j = 0; j < 8; ++j) am[j]     = (_Float16)sVals[koff + j][c];
        #pragma unroll
        for (int j = 0; j < 8; ++j) am[8 + j] = (_Float16)sVals[16 + koff + j][c];

        v8f acc = {0.f, 0.f, 0.f, 0.f, 0.f, 0.f, 0.f, 0.f};
        acc = __builtin_amdgcn_wmma_f32_16x16x32_f16(false, am, false, bm,
                                                     (short)0, acc, false, false);
        #pragma unroll
        for (int j = 0; j < 8; ++j) {
            int crow = t * 16 + mbase + j;   // channel index
            partial[(size_t)q * 1024 + crow * 16 + col] = (_Float16)acc[j];
        }
    }
}

// ---------------------------------------------------------------------------
// Final GEMM: out[m,o] = (sum_k partial[m,k]*lw[o,k] + lb[o]) / 32
// M = 16384, K = 1024, N = 64. 8 waves/block, one 16-row strip per wave,
// 32 K-steps x 4 N-tiles of v_wmma_f32_16x16x32_f16 (128 WMMA per wave).
// ---------------------------------------------------------------------------
__global__ __launch_bounds__(256) void k_final_gemm(const _Float16* __restrict__ Ap,
                                                    const _Float16* __restrict__ Bp,
                                                    const float* __restrict__ lb,
                                                    float* __restrict__ out) {
    const int wave = threadIdx.x >> 5;
    const int lane = threadIdx.x & 31;
    const int m0   = (blockIdx.x * 8 + wave) * 16;
    const int row  = m0 + (lane & 15);
    const int koff = (lane & 16) ? 8 : 0;

    v8f acc[4];
    v8f z = {0.f, 0.f, 0.f, 0.f, 0.f, 0.f, 0.f, 0.f};
    #pragma unroll
    for (int nt = 0; nt < 4; ++nt) acc[nt] = z;

    const _Float16* arow = Ap + (size_t)row * 1024;
    for (int kk = 0; kk < 32; ++kk) {
        const _Float16* ap = arow + kk * 32;
        v8h alo = *(const v8h*)(ap + koff);
        v8h ahi = *(const v8h*)(ap + 16 + koff);
        v16h am = __builtin_shufflevector(alo, ahi,
                     0, 1, 2, 3, 4, 5, 6, 7, 8, 9, 10, 11, 12, 13, 14, 15);
        #pragma unroll
        for (int nt = 0; nt < 4; ++nt) {
            const _Float16* bp = Bp + (((size_t)(kk * 4 + nt) * 32 + lane) * 16);
            v8h blo = *(const v8h*)bp;
            v8h bhi = *(const v8h*)(bp + 8);
            v16h bm = __builtin_shufflevector(blo, bhi,
                         0, 1, 2, 3, 4, 5, 6, 7, 8, 9, 10, 11, 12, 13, 14, 15);
            acc[nt] = __builtin_amdgcn_wmma_f32_16x16x32_f16(false, am, false, bm,
                                                             (short)0, acc[nt], false, false);
        }
    }

    const int mbase = (lane & 16) ? 8 : 0;
    const int ncol  = lane & 15;
    #pragma unroll
    for (int nt = 0; nt < 4; ++nt) {
        const int o = nt * 16 + ncol;
        const float bias = lb[o];
        #pragma unroll
        for (int j = 0; j < 8; ++j) {
            const int m = m0 + mbase + j;
            out[(size_t)m * CHOUT + o] = (acc[nt][j] + bias) * (1.0f / 32.0f);
        }
    }
}

// ---------------------------------------------------------------------------
extern "C" void kernel_launch(void* const* d_in, const int* in_sizes, int n_in,
                              void* d_out, int out_size, void* d_ws, size_t ws_size,
                              hipStream_t stream) {
    const float* inp_xyz   = (const float*)d_in[0];
    const float* inp_vals  = (const float*)d_in[1];
    const float* query_xyz = (const float*)d_in[2];
    const float* w1 = (const float*)d_in[3];
    const float* b1 = (const float*)d_in[4];
    const float* g1 = (const float*)d_in[5];
    const float* be1 = (const float*)d_in[6];
    const float* w2 = (const float*)d_in[7];
    const float* b2 = (const float*)d_in[8];
    const float* g2 = (const float*)d_in[9];
    const float* be2 = (const float*)d_in[10];
    const float* w3 = (const float*)d_in[11];
    const float* b3 = (const float*)d_in[12];
    const float* g3 = (const float*)d_in[13];
    const float* be3 = (const float*)d_in[14];
    const float* lw = (const float*)d_in[15];
    const float* lb = (const float*)d_in[16];
    float* out = (float*)d_out;

    char* ws = (char*)d_ws;
    int*      idx     = (int*)(ws + OFF_IDX);
    float*    stats   = (float*)(ws + OFF_STATS);
    float*    ss      = (float*)(ws + OFF_SS);
    _Float16* H3      = (_Float16*)(ws + OFF_H3);
    _Float16* partial = (_Float16*)(ws + OFF_PARTIAL);
    _Float16* bpack   = (_Float16*)(ws + OFF_BPACK);

    const int NQ = BB * SS;                 // 16384 queries

    k_init_stats<<<1, 512, 0, stream>>>(stats);
    k_knn<<<NQ, 32, 0, stream>>>(inp_xyz, query_xyz, idx);

    // WeightNet with training-mode BN: pass -> finalize, three times
    k_wn<<<PTOT / 256, 256, 0, stream>>>(1, idx, inp_xyz, query_xyz,
                                         w1, b1, w2, b2, w3, b3, ss, stats, H3);
    k_finalize<<<1, 64, 0, stream>>>(stats + 0,   ss + 0,   g1, be1, 32);
    k_wn<<<PTOT / 256, 256, 0, stream>>>(2, idx, inp_xyz, query_xyz,
                                         w1, b1, w2, b2, w3, b3, ss, stats, H3);
    k_finalize<<<1, 64, 0, stream>>>(stats + 64,  ss + 64,  g2, be2, 32);
    k_wn<<<PTOT / 256, 256, 0, stream>>>(3, idx, inp_xyz, query_xyz,
                                         w1, b1, w2, b2, w3, b3, ss, stats, H3);
    k_finalize<<<1, 64, 0, stream>>>(stats + 128, ss + 128, g3, be3, 16);
    k_wn<<<PTOT / 256, 256, 0, stream>>>(4, idx, inp_xyz, query_xyz,
                                         w1, b1, w2, b2, w3, b3, ss, stats, H3);

    k_pack_lw<<<65536 / 256, 256, 0, stream>>>(lw, bpack);
    k_aggregate<<<NQ, 32, 0, stream>>>(idx, inp_vals, H3, partial);
    k_final_gemm<<<NQ / (16 * 8), 256, 0, stream>>>(partial, bpack, lb, out);
}
